// TensorProductWithActivation_2491081032270
// MI455X (gfx1250) — matrix operand order repaired and checked by reading, not compiled, not run on hardware
//
#include <hip/hip_runtime.h>

typedef __attribute__((ext_vector_type(16))) __bf16 v16bf;
typedef __attribute__((ext_vector_type(8)))  float  v8f;
typedef __attribute__((ext_vector_type(4)))  float  f4x;
typedef __attribute__((ext_vector_type(4)))  int    i4x;

constexpr float INV_SQRT3 = 0.57735026918962576f;
constexpr float A0C = 0.0625f;                 // sqrt(1/256)
constexpr float A1C = 0.10825317547305482f;    // sqrt(3/256)
constexpr float C_SILU = 1.18544f;             // 1/sqrt(E[silu(z)^2])
constexpr float C_SIGM = 1.84623f;             // 1/sqrt(E[sigmoid(z)^2])

// ---- LDS layout (bytes) ----
#define OFF_BW12 0                 // bf16 [256][256]  (B[n][k], k = [a0cat])
#define OFF_BW3  131072            // bf16 [128][128]  (B[n][k] = W3[k][n])
#define OFF_BW4  163840            // bf16 [128][128]  (B[n][k] = W4[k][n])
#define OFF_VEC  196608            // per-wave float[16][384], stride 24576
#define OFF_Y    294912            // per-wave float[16][4],  stride 256
#define LDS_TOTAL (294912 + 4 * 256)

// ---- CDNA5 async LDS<->global paths (guarded; fallback = plain copies) ----
#if defined(__has_builtin)
#  if __has_builtin(__builtin_amdgcn_global_load_async_to_lds_b128)
#    define HAVE_ASYNC_LD 1
#  endif
#  if __has_builtin(__builtin_amdgcn_global_store_async_from_lds_b128)
#    define HAVE_ASYNC_ST 1
#  endif
#endif

#if defined(HAVE_ASYNC_LD) || defined(HAVE_ASYNC_ST)
#  if __has_builtin(__builtin_amdgcn_s_wait_asynccnt)
#    define WAIT_ASYNC0() __builtin_amdgcn_s_wait_asynccnt(0)
#  else
#    define WAIT_ASYNC0() asm volatile("s_wait_asynccnt 0x0" ::: "memory")
#  endif
#endif

// builtin signature (from compiler diagnostic): params are int4* (generic)
#define ASYNC_PTR(p) ((i4x*)(p))

// ---------------- weight prep: f32 -> bf16, transposed ----------------
__global__ void prep_weights(const float* __restrict__ W1, const float* __restrict__ W2,
                             const float* __restrict__ W3, const float* __restrict__ W4,
                             __bf16* __restrict__ ws) {
  int i = blockIdx.x * blockDim.x + threadIdx.x;
  if (i < 256 * 256) {                       // BW12[w][u], u over [x0 | x1dot]
    int w = i >> 8, u = i & 255;
    float v = (u < 128) ? W1[u * 256 + w] : W2[(u - 128) * 256 + w];
    ws[i] = (__bf16)v;
  } else if (i < 65536 + 16384) {            // BW3[w][u] = W3[u][w]
    int j = i - 65536; int w = j >> 7, u = j & 127;
    ws[i] = (__bf16)W3[u * 128 + w];
  } else if (i < 65536 + 32768) {            // BW4[w][u] = W4[u][w]
    int j = i - 65536 - 16384; int w = j >> 7, u = j & 127;
    ws[i] = (__bf16)W4[u * 128 + w];
  }
}

// ---------------- fragment builders ----------------
__device__ __forceinline__ void putX0(v16bf& a, int off, const float* row, int g, float scale) {
  const f4x* p = (const f4x*)(row + g);
  f4x q0 = p[0], q1 = p[1];
#pragma unroll
  for (int j = 0; j < 4; ++j) a[off + j]     = (__bf16)(q0[j] * scale);
#pragma unroll
  for (int j = 0; j < 4; ++j) a[off + 4 + j] = (__bf16)(q1[j] * scale);
}

__device__ __forceinline__ void putDot(v16bf& a, int off, const float* row, int u0,
                                       float ya, float yb, float yc) {
  union { f4x v[6]; float f[24]; } t;
  const f4x* p = (const f4x*)(row + 128 + 3 * u0);
#pragma unroll
  for (int j = 0; j < 6; ++j) t.v[j] = p[j];
#pragma unroll
  for (int j = 0; j < 8; ++j) {
    float d = t.f[3 * j] * ya + t.f[3 * j + 1] * yb + t.f[3 * j + 2] * yc;
    a[off + j] = (__bf16)(d * INV_SQRT3);
  }
}

__device__ __forceinline__ void putPick(v16bf& a, int off, const float* row, int u0,
                                        int k, float scale) {
  union { f4x v[6]; float f[24]; } t;
  const f4x* p = (const f4x*)(row + 128 + 3 * u0);
#pragma unroll
  for (int j = 0; j < 6; ++j) t.v[j] = p[j];
#pragma unroll
  for (int j = 0; j < 8; ++j) a[off + j] = (__bf16)(t.f[3 * j + k] * scale);
}

__device__ __forceinline__ v16bf load_bfrag(const __bf16* base, int n, int klen, int koff) {
  union { v16bf v; f4x q[2]; } u;
  const int h = (koff >> 4) & 1;
  const __bf16* p = base + (size_t)n * klen + (koff - 16 * h) + 8 * h;
  u.q[0] = *(const f4x*)p; u.q[1] = *(const f4x*)(p + 16);
  return u.v;
}

// ---------------- main kernel ----------------
extern "C" __global__ void __launch_bounds__(128, 1)
tp_main(const float* __restrict__ d1, const float* __restrict__ d2,
        const __bf16* __restrict__ wt, float* __restrict__ out,
        int numTiles, int waveStride) {
  extern __shared__ char smem[];
  __bf16* bw12 = (__bf16*)(smem + OFF_BW12);
  __bf16* bw3  = (__bf16*)(smem + OFF_BW3);
  __bf16* bw4  = (__bf16*)(smem + OFF_BW4);

  const int tid  = threadIdx.x;
  const int lane = tid & 31;
  const int wave = tid >> 5;
  const int hh   = lane >> 4;   // lane half (K-split select)
  const int mrow = lane & 15;   // A row / B column within tile
  float* vst = (float*)(smem + OFF_VEC + wave * 24576);
  float* yst = (float*)(smem + OFF_Y  + wave * 256);

  // stage all weights (192 KB bf16) into LDS once per workgroup
#ifdef HAVE_ASYNC_LD
  {
    char* src = (char*)wt;
    for (int i = tid; i < 12288; i += 128)
      __builtin_amdgcn_global_load_async_to_lds_b128(
          ASYNC_PTR(src + (size_t)i * 16), ASYNC_PTR(smem + (size_t)i * 16), 0, 0);
    WAIT_ASYNC0();
  }
#else
  {
    const f4x* src = (const f4x*)wt;
    f4x* dst = (f4x*)smem;
    for (int i = tid; i < 12288; i += 128) dst[i] = src[i];
  }
#endif
  __syncthreads();

  for (int tile = blockIdx.x * 4 + wave; tile < numTiles; tile += waveStride) {
#ifdef HAVE_ASYNC_ST
    WAIT_ASYNC0();   // vec staging buffer of previous tile fully drained
#endif
    const int tb = tile * 16;
    const float* row1 = d1 + (size_t)(tb + mrow) * 512;
    const f4x yv = *(const f4x*)(d2 + (size_t)(tb + mrow) * 4);
    const float y0 = yv[0], y1a = yv[1], y1b = yv[2], y1c = yv[3];
    if (lane < 16) *(f4x*)(yst + lane * 4) = yv;

    // prefetch next tile's activation rows (speculative, dropped if OOB)
    {
      int tile2 = tile + waveStride;
      if (tile2 < numTiles) {
        const char* pf = (const char*)(d1 + (size_t)(tile2 * 16 + mrow) * 512);
        __builtin_prefetch(pf, 0, 1);
        __builtin_prefetch(pf + 1024, 0, 1);
      }
    }

    // ---- A0cat fragments: k<128 -> x0*y0 ; k>=128 -> (x1 . y1)/sqrt3 ----
    v16bf af[8];
#pragma unroll
    for (int s = 0; s < 4; ++s) {
      int gA = s * 32 + hh * 8;
      putX0(af[s], 0, row1, gA, y0);
      putX0(af[s], 8, row1, gA + 16, y0);
    }
#pragma unroll
    for (int s = 4; s < 8; ++s) {
      int uA = s * 32 - 128 + hh * 8;
      putDot(af[s], 0, row1, uA, y1a, y1b, y1c);
      putDot(af[s], 8, row1, uA + 16, y1a, y1b, y1c);
    }

    float* orow = out + (size_t)tb * 512;

    // ---- phase 1a: scalar half of out0 -> silu, store directly ----
    for (int nt = 0; nt < 8; ++nt) {
      v8f acc = {};
#pragma unroll
      for (int s = 0; s < 8; ++s) {
        v16bf bf = load_bfrag(bw12, nt * 16 + mrow, 256, s * 32 + hh * 16);
        acc = __builtin_amdgcn_wmma_f32_16x16x32_bf16(false, af[s], false, bf,
                                                      (short)0, acc, false, false);
      }
#pragma unroll
      for (int v = 0; v < 8; ++v) {
        float x  = A0C * acc[v];
        float sg = 1.0f / (1.0f + __expf(-x));
        orow[(size_t)(v + hh * 8) * 512 + nt * 16 + mrow] = C_SILU * x * sg;
      }
    }

    // ---- phase 1b: gate half of out0 -> sigmoid, keep in registers ----
    v8f garr[8];
#pragma unroll
    for (int nt = 0; nt < 8; ++nt) {
      v8f acc = {};
#pragma unroll
      for (int s = 0; s < 8; ++s) {
        v16bf bf = load_bfrag(bw12, (nt + 8) * 16 + mrow, 256, s * 32 + hh * 16);
        acc = __builtin_amdgcn_wmma_f32_16x16x32_bf16(false, af[s], false, bf,
                                                      (short)0, acc, false, false);
      }
#pragma unroll
      for (int v = 0; v < 8; ++v) {
        float x = A0C * acc[v];
        garr[nt][v] = C_SIGM / (1.0f + __expf(-x));
      }
    }

    // ---- phase 2: P = x0 @ W3 (kept in registers, same C/D layout) ----
    v16bf xf[4];
#pragma unroll
    for (int s = 0; s < 4; ++s) {
      int gA = s * 32 + hh * 8;
      putX0(xf[s], 0, row1, gA, 1.0f);
      putX0(xf[s], 8, row1, gA + 16, 1.0f);
    }
    v8f Pv[8];
#pragma unroll
    for (int nt = 0; nt < 8; ++nt) {
      v8f acc = {};
#pragma unroll
      for (int s = 0; s < 4; ++s) {
        v16bf bf = load_bfrag(bw3, nt * 16 + mrow, 128, s * 32 + hh * 16);
        acc = __builtin_amdgcn_wmma_f32_16x16x32_bf16(false, xf[s], false, bf,
                                                      (short)0, acc, false, false);
      }
      Pv[nt] = acc;
    }

    // ---- phase 3: per k: Q_k = (x1[:,:,k]*y0/sqrt3) @ W4, fuse epilogue ----
    const float zscale = y0 * INV_SQRT3;
#pragma unroll
    for (int k = 0; k < 3; ++k) {
      v16bf zf[4];
#pragma unroll
      for (int s = 0; s < 4; ++s) {
        int uA = s * 32 + hh * 8;
        putPick(zf[s], 0, row1, uA, k, zscale);
        putPick(zf[s], 8, row1, uA + 16, k, zscale);
      }
      float ycol[8];
#pragma unroll
      for (int v = 0; v < 8; ++v) ycol[v] = yst[(v + hh * 8) * 4 + 1 + k];
#pragma unroll
      for (int nt = 0; nt < 8; ++nt) {
        v8f acc = {};
#pragma unroll
        for (int s = 0; s < 4; ++s) {
          v16bf bf = load_bfrag(bw4, nt * 16 + mrow, 128, s * 32 + hh * 16);
          acc = __builtin_amdgcn_wmma_f32_16x16x32_bf16(false, zf[s], false, bf,
                                                        (short)0, acc, false, false);
        }
#pragma unroll
        for (int v = 0; v < 8; ++v) {
          float o1 = A1C * (Pv[nt][v] * ycol[v] * INV_SQRT3 + acc[v]);
          int r = v + hh * 8;
          int w = nt * 16 + mrow;
          vst[r * 384 + 3 * w + k] = garr[nt][v] * o1;
        }
      }
    }

    // ---- flush vec staging buffer -> global ----
#ifdef HAVE_ASYNC_ST
    asm volatile("s_wait_dscnt 0x0" ::: "memory");  // vst writes visible to async engine
#pragma unroll 4
    for (int r = 0; r < 16; ++r) {
      char* g = (char*)(orow + (size_t)r * 512 + 128 + lane * 12);
      char* l = (char*)(vst + r * 384 + lane * 12);
#pragma unroll
      for (int c = 0; c < 3; ++c)
        __builtin_amdgcn_global_store_async_from_lds_b128(
            ASYNC_PTR(g + c * 16), ASYNC_PTR(l + c * 16), 0, 0);
    }
#else
#pragma unroll 4
    for (int r = 0; r < 16; ++r) {
      const f4x* src = (const f4x*)(vst + r * 384 + lane * 12);
      f4x* dstp = (f4x*)(orow + (size_t)r * 512 + 128 + lane * 12);
      dstp[0] = src[0]; dstp[1] = src[1]; dstp[2] = src[2];
    }
#endif
  }
  // s_endpgm performs an implicit wait-idle, draining any outstanding async stores
}

// ---------------- launcher ----------------
extern "C" void kernel_launch(void* const* d_in, const int* in_sizes, int n_in,
                              void* d_out, int out_size, void* d_ws, size_t ws_size,
                              hipStream_t stream) {
  const float* d1 = (const float*)d_in[0];
  const float* d2 = (const float*)d_in[1];
  const float* W1 = (const float*)d_in[2];
  const float* W2 = (const float*)d_in[3];
  const float* W3 = (const float*)d_in[4];
  const float* W4 = (const float*)d_in[5];
  float* out = (float*)d_out;
  __bf16* wt = (__bf16*)d_ws;   // 196608 bytes of bf16 weights

  prep_weights<<<(98304 + 255) / 256, 256, 0, stream>>>(W1, W2, W3, W4, wt);

  int n = in_sizes[0] / 512;
  int numTiles = n / 16;
  int numWG = 256;
  if (numWG * 4 > numTiles) numWG = (numTiles + 3) / 4;
  tp_main<<<numWG, 128, LDS_TOTAL, stream>>>(d1, d2, wt, out, numTiles, numWG * 4);
}
